// Attention_59571196395684
// MI455X (gfx1250) — matrix operand order repaired and checked by reading, not compiled, not run on hardware
//
#include <hip/hip_runtime.h>
#include <math.h>

// ---------------------------------------------------------------------------
// MI455X (gfx1250) fused transposed-attention block.
//
// Roofline: ~123 GFLOP of GEMM vs ~400MB of unavoidable HBM traffic
// (x in, out out) => memory bound (~17us at 23.3 TB/s) IF intermediates stay
// in the 192MB L2. Pipeline runs per-batch (50MB kv slab + 25MB v slab stay
// L2-resident between kernels); attn@v is algebraically fused with the
// projection GEMM so "out" (402MB) is never materialized.
//
// GEMMs: v_wmma_f32_16x16x32_f16 with an f16 hi/lo split (3 WMMAs per K-step,
// ~2^-22 relative error). LDS staging stores hi/lo as k-pair-packed u32 so
// every fragment VGPR is filled by contiguous LDS b128 loads (no 16-bit
// shuffles). A-operands (kv_w, folded attn*proj matrix) are pre-packed in
// global memory since they are tiny and reused by all 128 n-tiles.
// ---------------------------------------------------------------------------

typedef _Float16 v16h __attribute__((ext_vector_type(16)));
typedef _Float16 v2h  __attribute__((ext_vector_type(2)));
typedef float    v8f  __attribute__((ext_vector_type(8)));

union HFrag { v16h v; unsigned u[8]; uint4 q[2]; };
union FAcc  { v8f  v; float f[8]; };
union PkU   { v2h  h; unsigned u; };

#define HW    16384
#define KDIM  384

// split a,b into f16 hi + f16 residual-lo; pack pairs into u32 (a in [15:0]).
__device__ __forceinline__ void split_pack(float a, float b,
                                           unsigned& hi, unsigned& lo) {
  _Float16 ha = (_Float16)a, hb = (_Float16)b;
  float ra = a - (float)ha, rb = b - (float)hb;
  PkU H, L;
  H.h = v2h{ha, hb};
  L.h = v2h{(_Float16)ra, (_Float16)rb};
  hi = H.u;
  lo = L.u;
}

// ---------------------------------------------------------------------------
// Prepass: pack an [M x 384] f32 A-matrix into k-pair u32 hi/lo arrays
// [M x 192]. total = M*192.
// ---------------------------------------------------------------------------
__global__ __launch_bounds__(256) void pack_a_kernel(
    const float* __restrict__ A, unsigned* __restrict__ Aph,
    unsigned* __restrict__ Apl, int total)
{
  int i = blockIdx.x * 256 + threadIdx.x;
  if (i >= total) return;
  unsigned hi, lo;
  split_pack(A[2 * i], A[2 * i + 1], hi, lo);
  Aph[i] = hi;
  Apl[i] = lo;
}

// ---------------------------------------------------------------------------
// C[o,n] = sum_k A[o,k]*B[k,n] + bias[o]
//   A: pre-packed hi/lo u32 [Mrows x 192] (k-pairs), B: f32 [384 x 16384].
//   grid = (16384/128, Mrows/128), block = 256 (8 waves).
//   Wave tile: 32(o) x 64(n) = 2x4 tiles of 16x16.
// ---------------------------------------------------------------------------
__global__ __launch_bounds__(256) void gemm_bias_wmma(
    const unsigned* __restrict__ Aph,
    const unsigned* __restrict__ Apl,
    const float*    __restrict__ B,
    const float*    __restrict__ bias,
    float*          __restrict__ C)
{
  // [row][k2] u32 (k-pair packed f16). 20 u32/row: 16B-aligned rows, no
  // bank conflicts across lanes 0-15 on b128 fragment loads.
  __shared__ __align__(16) unsigned AhU[128][20], AlU[128][20];
  __shared__ __align__(16) unsigned BhU[128][20], BlU[128][20];

  const int tid  = threadIdx.x;
  const int wave = tid >> 5;
  const int lane = tid & 31;
  const int lhi  = lane >> 4;
  const int llo  = lane & 15;
  const int n0   = blockIdx.x << 7;
  const int o0   = blockIdx.y << 7;
  const int osub = (wave >> 1) << 5;   // 0,32,64,96
  const int nsub = (wave & 1) << 6;    // 0,64

  FAcc acc[2][4];
  #pragma unroll
  for (int oi = 0; oi < 2; ++oi)
    #pragma unroll
    for (int t = 0; t < 4; ++t)
      #pragma unroll
      for (int r = 0; r < 8; ++r) acc[oi][t].f[r] = 0.0f;

  for (int k0 = 0; k0 < KDIM; k0 += 32) {
    const int k2base = k0 >> 1;
    // --- stage A: pure u32 copy of 128(o) x 16(k2), uint4 loads/stores ---
    #pragma unroll
    for (int i = 0; i < 2; ++i) {
      int idx = tid + (i << 8);            // 0..511
      int ol  = idx >> 2;                  // 0..127
      int k4  = (idx & 3) << 2;            // 0,4,8,12
      size_t g = (size_t)(o0 + ol) * 192 + k2base + k4;
      *reinterpret_cast<uint4*>(&AhU[ol][k4]) =
          *reinterpret_cast<const uint4*>(Aph + g);
      *reinterpret_cast<uint4*>(&AlU[ol][k4]) =
          *reinterpret_cast<const uint4*>(Apl + g);
    }
    // --- stage B: convert f32 -> f16 hi/lo, store transposed [n][k2] ---
    #pragma unroll
    for (int i = 0; i < 8; ++i) {
      int idx = tid + (i << 8);            // 0..2047
      int nl  = idx & 127;
      int kk2 = idx >> 7;                  // 0..15
      const float* bp = B + (size_t)(k0 + (kk2 << 1)) * HW + n0 + nl;
      float b0 = bp[0];
      float b1 = bp[HW];
      unsigned hi, lo;
      split_pack(b0, b1, hi, lo);
      BhU[nl][kk2] = hi;
      BlU[nl][kk2] = lo;
    }
    __syncthreads();

    // A fragments: u32 cols {4*lhi+0..3, 4*lhi+8..11}  (ISA 16-bit A 16x32)
    HFrag ah[2], al[2];
    #pragma unroll
    for (int oi = 0; oi < 2; ++oi) {
      const int orow = osub + (oi << 4) + llo;
      ah[oi].q[0] = *reinterpret_cast<const uint4*>(&AhU[orow][lhi << 2]);
      ah[oi].q[1] = *reinterpret_cast<const uint4*>(&AhU[orow][(lhi << 2) + 8]);
      al[oi].q[0] = *reinterpret_cast<const uint4*>(&AlU[orow][lhi << 2]);
      al[oi].q[1] = *reinterpret_cast<const uint4*>(&AlU[orow][(lhi << 2) + 8]);
    }
    #pragma unroll
    for (int t = 0; t < 4; ++t) {
      // B fragment: u32 cols {4*lhi+0..3, 4*lhi+8..11}  (same k order as A)
      const int nrow = nsub + (t << 4) + llo;
      HFrag bh, bl;
      bh.q[0] = *reinterpret_cast<const uint4*>(&BhU[nrow][lhi << 2]);
      bh.q[1] = *reinterpret_cast<const uint4*>(&BhU[nrow][(lhi << 2) + 8]);
      bl.q[0] = *reinterpret_cast<const uint4*>(&BlU[nrow][lhi << 2]);
      bl.q[1] = *reinterpret_cast<const uint4*>(&BlU[nrow][(lhi << 2) + 8]);
      #pragma unroll
      for (int oi = 0; oi < 2; ++oi) {
        // f16x3: hi*hi + hi*lo + lo*hi, f32 accumulate
        acc[oi][t].v = __builtin_amdgcn_wmma_f32_16x16x32_f16(
            false, ah[oi].v, false, bh.v, (short)0, acc[oi][t].v, false, false);
        acc[oi][t].v = __builtin_amdgcn_wmma_f32_16x16x32_f16(
            false, ah[oi].v, false, bl.v, (short)0, acc[oi][t].v, false, false);
        acc[oi][t].v = __builtin_amdgcn_wmma_f32_16x16x32_f16(
            false, al[oi].v, false, bh.v, (short)0, acc[oi][t].v, false, false);
      }
    }
    __syncthreads();
  }

  // C/D layout: VGPR r -> M = r + 8*lhi, N = llo
  #pragma unroll
  for (int oi = 0; oi < 2; ++oi) {
    float bv_[8];
    #pragma unroll
    for (int r = 0; r < 8; ++r)
      bv_[r] = bias[o0 + osub + (oi << 4) + r + (lhi << 3)];
    #pragma unroll
    for (int t = 0; t < 4; ++t) {
      #pragma unroll
      for (int r = 0; r < 8; ++r) {
        int o = o0 + osub + (oi << 4) + r + (lhi << 3);
        int n = n0 + nsub + (t << 4) + llo;
        C[(size_t)o * HW + n] = acc[oi][t].f[r] + bv_[r];
      }
    }
  }
}

// ---------------------------------------------------------------------------
// K2: depthwise 3x3 ('SAME') on kv0[768][128][128] + bias.
//   grid = (768 channels, 2 half-images); half-image + halo staged in LDS.
//   v channels -> vbuf; k channels -> 48 segment sums + sum(k^2) only.
// ---------------------------------------------------------------------------
__global__ __launch_bounds__(256) void dw3x3_kernel(
    const float* __restrict__ kv0,
    const float* __restrict__ dw_w,
    const float* __restrict__ dw_b,
    float* __restrict__ vbuf,         // [384][16384]
    float* __restrict__ kseg,         // [2][384][48]
    float* __restrict__ sumsq)        // [2][384]
{
  __shared__ float img[66 * 128];
  __shared__ float segAcc[48];
  __shared__ float ssAcc;

  const int tid  = threadIdx.x;
  const int ch   = blockIdx.x;
  const int half = blockIdx.y;
  const int y0   = half << 6;
  const float* src = kv0 + (size_t)ch * HW;

  for (int idx = tid; idx < 66 * 128; idx += 256) {
    int gy = y0 - 1 + (idx >> 7);
    img[idx] = (gy >= 0 && gy < 128) ? src[(gy << 7) + (idx & 127)] : 0.0f;
  }
  float w[9];
  #pragma unroll
  for (int j = 0; j < 9; ++j) w[j] = dw_w[ch * 9 + j];
  const float bias = dw_b[ch];
  const bool  isK  = (ch < 384);
  if (tid < 48) segAcc[tid] = 0.0f;
  if (tid == 0) ssAcc = 0.0f;
  __syncthreads();

  float ss = 0.0f;
  for (int s = 0; s < 32; ++s) {
    int nl = tid + (s << 8);
    int ly = nl >> 7;
    int lx = nl & 127;
    float acc = bias;
    #pragma unroll
    for (int dy = 0; dy < 3; ++dy) {
      const float* rp = &img[(ly + dy) << 7];
      #pragma unroll
      for (int dx = -1; dx <= 1; ++dx) {
        int xx = lx + dx;
        float v = (xx >= 0 && xx < 128) ? rp[xx] : 0.0f;
        acc += w[dy * 3 + (dx + 1)] * v;
      }
    }
    int n = ((y0 + ly) << 7) + lx;
    if (isK) {
      atomicAdd(&segAcc[(n * 3) >> 10], acc);   // seg(n) = n*48/16384
      ss += acc * acc;
    } else {
      vbuf[(size_t)(ch - 384) * HW + n] = acc;
    }
  }
  if (isK) {
    atomicAdd(&ssAcc, ss);
    __syncthreads();
    if (tid < 48) kseg[((size_t)half * 384 + ch) * 48 + tid] = segAcc[tid];
    if (tid == 0) sumsq[half * 384 + ch] = ssAcc;
  }
}

// ---------------------------------------------------------------------------
// K3: per head: attn = softmax(temp * q_src @ Kseg^T / ||k||); fold proj_w:
//   M[o, h*48+d] = sum_c proj_w[o, h*48+c] * attn[c, d], emitted pre-packed
//   (f16 hi/lo k-pair u32) as the A-operand of the final GEMM.
//   grid = 8 heads, block = 256.
// ---------------------------------------------------------------------------
__global__ __launch_bounds__(256) void attn_proj_kernel(
    const float* __restrict__ kseg,        // [2][384][48]
    const float* __restrict__ sumsq,       // [2][384]
    const float* __restrict__ q_param,     // [8][48][48]
    const float* __restrict__ temperature, // [8]
    const float* __restrict__ proj_w,      // [384][384]
    unsigned* __restrict__ Mph,            // [384][192]
    unsigned* __restrict__ Mpl)            // [384][192]
{
  __shared__ float ks[48][48];
  __shared__ float qs[48][48];
  __shared__ float at[48][48];
  __shared__ float invn[48];

  const int h   = blockIdx.x;
  const int tid = threadIdx.x;

  for (int e = tid; e < 48 * 48; e += 256) {
    int d = e / 48, j = e % 48;
    int chd = h * 48 + d;
    ks[d][j] = kseg[(size_t)chd * 48 + j] + kseg[(size_t)(384 + chd) * 48 + j];
    qs[d][j] = q_param[(size_t)h * 48 * 48 + e];
  }
  if (tid < 48) {
    int chd = h * 48 + tid;
    float nrm = sqrtf(sumsq[chd] + sumsq[384 + chd]);
    invn[tid] = temperature[h] / fmaxf(nrm, 1e-12f);
  }
  __syncthreads();

  for (int e = tid; e < 48 * 48; e += 256) {
    int c = e / 48, d = e % 48;
    float s = 0.0f;
    #pragma unroll 8
    for (int j = 0; j < 48; ++j) s += qs[c][j] * ks[d][j];
    at[c][d] = s * invn[d];
  }
  __syncthreads();

  if (tid < 48) {
    float m = -1e30f;
    for (int d = 0; d < 48; ++d) m = fmaxf(m, at[tid][d]);
    float sum = 0.0f;
    for (int d = 0; d < 48; ++d) sum += expf(at[tid][d] - m);
    float inv = 1.0f / sum;
    for (int d = 0; d < 48; ++d) at[tid][d] = expf(at[tid][d] - m) * inv;
  }
  __syncthreads();

  for (int e = tid; e < 384 * 24; e += 256) {   // d-pairs: pack directly
    int o = e / 24, d2 = e % 24;
    const float* pw = proj_w + (size_t)o * 384 + h * 48;
    float s0 = 0.0f, s1 = 0.0f;
    #pragma unroll 8
    for (int c = 0; c < 48; ++c) {
      float a = pw[c];
      s0 += a * at[c][2 * d2];
      s1 += a * at[c][2 * d2 + 1];
    }
    unsigned hi, lo;
    split_pack(s0, s1, hi, lo);
    size_t k2 = (size_t)o * 192 + h * 24 + d2;
    Mph[k2] = hi;
    Mpl[k2] = lo;
  }
}

// ---------------------------------------------------------------------------
// Host-side launch: per-batch pipeline keeps intermediates L2-resident.
// ---------------------------------------------------------------------------
extern "C" void kernel_launch(void* const* d_in, const int* in_sizes, int n_in,
                              void* d_out, int out_size, void* d_ws, size_t ws_size,
                              hipStream_t stream) {
  (void)in_sizes; (void)n_in; (void)out_size; (void)ws_size;
  const float* x           = (const float*)d_in[0];
  const float* q_param     = (const float*)d_in[1];
  const float* temperature = (const float*)d_in[2];
  const float* kv_w        = (const float*)d_in[3];
  const float* kv_b        = (const float*)d_in[4];
  const float* dw_w        = (const float*)d_in[5];
  const float* dw_b        = (const float*)d_in[6];
  const float* proj_w      = (const float*)d_in[7];
  const float* proj_b      = (const float*)d_in[8];
  float*       out         = (float*)d_out;

  float* ws    = (float*)d_ws;
  float* kv0   = ws;                          // 768*16384 f32 (50.3 MB)
  float* vbuf  = kv0  + (size_t)768 * HW;     // 384*16384 f32 (25.2 MB)
  float* kseg  = vbuf + (size_t)384 * HW;     // 2*384*48
  float* sumsq = kseg + 2 * 384 * 48;         // 2*384
  unsigned* Wph = (unsigned*)(sumsq + 2 * 384);   // kv_w packed hi [768*192]
  unsigned* Wpl = Wph + 768 * 192;                // kv_w packed lo
  unsigned* Mph = Wpl + 768 * 192;                // folded attn*proj hi [384*192]
  unsigned* Mpl = Mph + 384 * 192;                // folded attn*proj lo

  // Pack kv_w once (reused by all 8 batches).
  pack_a_kernel<<<(768 * 192 + 255) / 256, 256, 0, stream>>>(kv_w, Wph, Wpl,
                                                             768 * 192);

  for (int b = 0; b < 8; ++b) {
    const float* xb   = x   + (size_t)b * KDIM * HW;
    float*       outb = out + (size_t)b * KDIM * HW;

    // K1: kv0 = kv_w @ x_b + kv_b     (768 x 384 x 16384 WMMA GEMM)
    gemm_bias_wmma<<<dim3(HW / 128, 768 / 128), 256, 0, stream>>>(
        Wph, Wpl, xb, kv_b, kv0);

    // K2: depthwise 3x3; emit v, Kseg, sum(k^2)
    dw3x3_kernel<<<dim3(768, 2), 256, 0, stream>>>(kv0, dw_w, dw_b, vbuf,
                                                   kseg, sumsq);

    // K3: 48x48 attention + softmax per head, folded with proj_w -> packed M
    attn_proj_kernel<<<8, 256, 0, stream>>>(kseg, sumsq, q_param, temperature,
                                            proj_w, Mph, Mpl);

    // K4: out_b = M @ v + proj_b      (384 x 384 x 16384 WMMA GEMM)
    gemm_bias_wmma<<<dim3(HW / 128, 384 / 128), 256, 0, stream>>>(
        Mph, Mpl, vbuf, proj_b, outb);
  }
}
